// MambaBlock_62792421867548
// MI455X (gfx1250) — compile-verified, matrix operand order checked by reading, not run on hardware
//
#include <hip/hip_runtime.h>
#include <hip/hip_bf16.h>

// ---------------------------------------------------------------------------
// Problem constants (from reference)
// ---------------------------------------------------------------------------
constexpr int kDIM    = 1024;
constexpr int kDSTATE = 16;
constexpr int kDCONV  = 4;
constexpr int kDINNER = 2048;   // DIM * EXPAND
constexpr int kDTRANK = 64;     // DIM / 16
constexpr int kBATCH  = 4;
constexpr int kT      = 2048;
constexpr int kBT     = kBATCH * kT;           // 8192 token rows
constexpr int kXDBL   = kDTRANK + 2 * kDSTATE; // 96

typedef __attribute__((ext_vector_type(16))) __bf16 v16bf;
typedef __attribute__((ext_vector_type(8)))  float  v8f;
typedef int v4i __attribute__((__vector_size__(4 * sizeof(int))));

// ---------------------------------------------------------------------------
// CDNA5 async global->LDS path (ASYNCcnt).  Gated on builtin availability;
// falls back to the proven sync global->VGPR->LDS staging otherwise.
// The builtin takes (v4i addrspace(1)*, v4i addrspace(3)*, imm offset, imm cpol).
// ---------------------------------------------------------------------------
#if defined(__has_builtin)
#if __has_builtin(__builtin_amdgcn_global_load_async_to_lds_b128) && \
    __has_builtin(__builtin_amdgcn_s_wait_asynccnt)
#define USE_ASYNC_LDS 1
#endif
#endif

#ifdef USE_ASYNC_LDS
template <int OFF>
__device__ __forceinline__ void async_copy_b128(const void* g, void* l)
{
    // INST_OFFSET is added to BOTH the global and LDS addresses (ISA 10.x),
    // so one base pair covers offset 0 and +16B.
    __builtin_amdgcn_global_load_async_to_lds_b128(
        (__attribute__((address_space(1))) v4i*)(g),
        (__attribute__((address_space(3))) v4i*)(l),
        OFF, 0);
}
#endif

// ---------------------------------------------------------------------------
// Generic bf16 WMMA GEMM:  C[M,N] = A[M,K] (bf16, row-major, lda)
//                                 x W[N,K]^T (bf16, row-major, ldw)
// Block tile 128(M) x 64(N), BK=32.  8 waves, each wave computes 32x32 as a
// 2x2 grid of v_wmma_f32_16x16x32_bf16.  Double-buffered LDS with async
// global->LDS copies when available (one barrier per K-step).
// Epilogue modes:
//   0: store f32 to Cf
//   1: softplus(c + bias[col]) -> Cf            (delta projection)
//   2: bf16(c) -> Cbf[row*ldc + col_off + col]  (build fwd||bwd concat)
//   3: (c + bias[col] + resid[row*ldc+col]) * mask[row] -> Cf (final output)
// ---------------------------------------------------------------------------
__global__ __launch_bounds__(256)
void gemm_bf16_wmma(const __hip_bfloat16* __restrict__ A, int lda,
                    const __hip_bfloat16* __restrict__ W, int ldw,
                    float* __restrict__ Cf, __hip_bfloat16* __restrict__ Cbf,
                    int ldc, int col_off,
                    int M, int N, int K, int mode,
                    const float* __restrict__ bias,
                    const float* __restrict__ resid,
                    const unsigned char* __restrict__ mask)
{
    // double-buffered tiles (24 KB total; fallback path uses buffer 0 only)
    __shared__ __align__(16) unsigned short As[2 * 128 * 32];
    __shared__ __align__(16) unsigned short Bs[2 * 64 * 32];

    const int tid  = threadIdx.x;
    const int lane = tid & 31;
    const int wave = tid >> 5;      // 0..7
    const int wm   = wave >> 1;     // 0..3 : 32-row strip inside block tile
    const int wn   = wave & 1;      // 0..1 : 32-col strip inside block tile
    const int m0   = blockIdx.y * 128;
    const int n0   = blockIdx.x * 64;

    v8f acc[2][2];
#pragma unroll
    for (int i = 0; i < 2; ++i)
#pragma unroll
        for (int j = 0; j < 2; ++j) acc[i][j] = v8f{};

    const unsigned short* Ag = (const unsigned short*)A;
    const unsigned short* Wg = (const unsigned short*)W;

    // staging assignments: A tile 128x32 (16 bf16/thread), B tile 64x32 (8/thread)
    const int ar = tid >> 1, ac = (tid & 1) << 4;
    const int br = tid >> 2, bc = (tid & 3) << 3;
    const int nrow = n0 + br;

    // fragment addressing (ISA 7.12.2, 16-bit A 16x32 layout; B mirrored)
    const int rsel = lane & 15;
    const int kb   = (lane < 16) ? 0 : 8;
    union FR { uint4 q[2]; v16bf v; };

#ifdef USE_ASYNC_LDS
    // ---- async double-buffered pipeline -----------------------------------
    // OOB B rows are never written by async copies: pre-zero them once.
    if (nrow >= N) {
        *(uint4*)&Bs[0 * 2048 + br * 32 + bc] = make_uint4(0u, 0u, 0u, 0u);
        *(uint4*)&Bs[1 * 2048 + br * 32 + bc] = make_uint4(0u, 0u, 0u, 0u);
    }
    __syncthreads();

    // prologue: issue K-tile 0 into buffer 0
    {
        const unsigned short* ap = Ag + (size_t)(m0 + ar) * lda + ac;
        unsigned short* al = &As[ar * 32 + ac];
        async_copy_b128<0>(ap, al);
        async_copy_b128<16>(ap, al);              // +16B = +8 bf16, both sides
        if (nrow < N)
            async_copy_b128<0>(Wg + (size_t)nrow * ldw + bc,
                               &Bs[br * 32 + bc]);
    }

    int buf = 0;
    for (int k0 = 0; k0 < K; k0 += 32) {
        __builtin_amdgcn_s_wait_asynccnt(0);  // current buffer's copies done
        __syncthreads();                      // visible to all waves; prev
                                              // compute done -> next buf free
        const int nb = buf ^ 1;
        if (k0 + 32 < K) {                    // issue next tile, overlap WMMA
            const unsigned short* ap =
                Ag + (size_t)(m0 + ar) * lda + (k0 + 32) + ac;
            unsigned short* al = &As[nb * 4096 + ar * 32 + ac];
            async_copy_b128<0>(ap, al);
            async_copy_b128<16>(ap, al);
            if (nrow < N)
                async_copy_b128<0>(Wg + (size_t)nrow * ldw + (k0 + 32) + bc,
                                   &Bs[nb * 2048 + br * 32 + bc]);
        }

        const unsigned short* AsB = As + buf * 4096;
        const unsigned short* BsB = Bs + buf * 2048;
        FR a0, a1, b0, b1;
        {
            const unsigned short* p = &AsB[(wm * 32 + rsel) * 32 + kb];
            a0.q[0] = *(const uint4*)p;  a0.q[1] = *(const uint4*)(p + 16);
            p = &AsB[(wm * 32 + 16 + rsel) * 32 + kb];
            a1.q[0] = *(const uint4*)p;  a1.q[1] = *(const uint4*)(p + 16);
            p = &BsB[(wn * 32 + rsel) * 32 + kb];
            b0.q[0] = *(const uint4*)p;  b0.q[1] = *(const uint4*)(p + 16);
            p = &BsB[(wn * 32 + 16 + rsel) * 32 + kb];
            b1.q[0] = *(const uint4*)p;  b1.q[1] = *(const uint4*)(p + 16);
        }
        acc[0][0] = __builtin_amdgcn_wmma_f32_16x16x32_bf16(false, a0.v, false, b0.v, (short)0, acc[0][0], false, false);
        acc[0][1] = __builtin_amdgcn_wmma_f32_16x16x32_bf16(false, a0.v, false, b1.v, (short)0, acc[0][1], false, false);
        acc[1][0] = __builtin_amdgcn_wmma_f32_16x16x32_bf16(false, a1.v, false, b0.v, (short)0, acc[1][0], false, false);
        acc[1][1] = __builtin_amdgcn_wmma_f32_16x16x32_bf16(false, a1.v, false, b1.v, (short)0, acc[1][1], false, false);
        buf = nb;
    }
#else
    // ---- fallback: sync staging, single buffer, two barriers per K-step ---
    for (int k0 = 0; k0 < K; k0 += 32) {
        const unsigned short* ap = Ag + (size_t)(m0 + ar) * lda + k0 + ac;
        uint4 av0 = *(const uint4*)ap;
        uint4 av1 = *(const uint4*)(ap + 8);
        *(uint4*)&As[ar * 32 + ac]     = av0;
        *(uint4*)&As[ar * 32 + ac + 8] = av1;
        uint4 bv = make_uint4(0u, 0u, 0u, 0u);
        const unsigned short* bp = Wg + (size_t)nrow * ldw + k0 + bc;
        if (nrow < N) bv = *(const uint4*)bp;
        *(uint4*)&Bs[br * 32 + bc] = bv;
        if (k0 + 32 < K) {
            __builtin_prefetch(ap + 32, 0, 1);
            if (nrow < N) __builtin_prefetch(bp + 32, 0, 1);
        }
        __syncthreads();

        FR a0, a1, b0, b1;
        {
            const unsigned short* p = &As[(wm * 32 + rsel) * 32 + kb];
            a0.q[0] = *(const uint4*)p;  a0.q[1] = *(const uint4*)(p + 16);
            p = &As[(wm * 32 + 16 + rsel) * 32 + kb];
            a1.q[0] = *(const uint4*)p;  a1.q[1] = *(const uint4*)(p + 16);
            p = &Bs[(wn * 32 + rsel) * 32 + kb];
            b0.q[0] = *(const uint4*)p;  b0.q[1] = *(const uint4*)(p + 16);
            p = &Bs[(wn * 32 + 16 + rsel) * 32 + kb];
            b1.q[0] = *(const uint4*)p;  b1.q[1] = *(const uint4*)(p + 16);
        }
        acc[0][0] = __builtin_amdgcn_wmma_f32_16x16x32_bf16(false, a0.v, false, b0.v, (short)0, acc[0][0], false, false);
        acc[0][1] = __builtin_amdgcn_wmma_f32_16x16x32_bf16(false, a0.v, false, b1.v, (short)0, acc[0][1], false, false);
        acc[1][0] = __builtin_amdgcn_wmma_f32_16x16x32_bf16(false, a1.v, false, b0.v, (short)0, acc[1][0], false, false);
        acc[1][1] = __builtin_amdgcn_wmma_f32_16x16x32_bf16(false, a1.v, false, b1.v, (short)0, acc[1][1], false, false);
        __syncthreads();
    }
#endif

    // ---- epilogue: C/D layout (ISA 7.12.2): VGPR r -> M = r (+8 for hi lanes)
    const int rowadd = (lane < 16) ? 0 : 8;
    const int colin  = lane & 15;
#pragma unroll
    for (int ti = 0; ti < 2; ++ti) {
#pragma unroll
        for (int tj = 0; tj < 2; ++tj) {
            const int mt   = m0 + wm * 32 + ti * 16;
            const int gcol = n0 + wn * 32 + tj * 16 + colin;
            if (gcol >= N) continue;
#pragma unroll
            for (int r = 0; r < 8; ++r) {
                const int grow = mt + r + rowadd;
                float v = acc[ti][tj][r];
                if (mode == 0) {
                    Cf[(size_t)grow * ldc + gcol] = v;
                } else if (mode == 1) {
                    v += bias[gcol];
                    v = (v > 20.0f) ? v : log1pf(__expf(v));   // softplus
                    Cf[(size_t)grow * ldc + gcol] = v;
                } else if (mode == 2) {
                    Cbf[(size_t)grow * ldc + col_off + gcol] = __float2bfloat16(v);
                } else {
                    v += bias[gcol] + resid[(size_t)grow * ldc + gcol];
                    v *= mask[grow] ? 1.0f : 0.0f;
                    Cf[(size_t)grow * ldc + gcol] = v;
                }
            }
        }
    }
}

// ---------------------------------------------------------------------------
// LayerNorm over DIM, emit bf16 (A-matrix input of the in-projection GEMM)
// ---------------------------------------------------------------------------
__global__ __launch_bounds__(256)
void layernorm_bf16(const float* __restrict__ x, const float* __restrict__ w,
                    const float* __restrict__ b, __hip_bfloat16* __restrict__ out)
{
    __shared__ float s1[256], s2[256];
    const int row = blockIdx.x;
    const float* xr = x + (size_t)row * kDIM;
    float sum = 0.f, sumsq = 0.f;
    for (int i = threadIdx.x; i < kDIM; i += 256) {
        float v = xr[i]; sum += v; sumsq += v * v;
    }
    s1[threadIdx.x] = sum; s2[threadIdx.x] = sumsq;
    __syncthreads();
    for (int s = 128; s > 0; s >>= 1) {
        if (threadIdx.x < s) { s1[threadIdx.x] += s1[threadIdx.x + s];
                               s2[threadIdx.x] += s2[threadIdx.x + s]; }
        __syncthreads();
    }
    const float mu   = s1[0] * (1.0f / kDIM);
    const float var  = s2[0] * (1.0f / kDIM) - mu * mu;
    const float rstd = rsqrtf(var + 1e-5f);
    __hip_bfloat16* orow = out + (size_t)row * kDIM;
    for (int i = threadIdx.x; i < kDIM; i += 256)
        orow[i] = __float2bfloat16((xr[i] - mu) * rstd * w[i] + b[i]);
}

// ---------------------------------------------------------------------------
// Causal depthwise conv (D_CONV=4) + SiLU.  rev=1 realizes the time-flipped
// branch without materializing a flipped sequence:
//   fwd: y[t] = sum_k w[k] * x[t-3+k]     bwd: y[t] = sum_k w[k] * x[t+3-k]
// Input xi_pre is the first D_INNER columns of xz [BT, 2*D_INNER].
// ---------------------------------------------------------------------------
__global__ void conv_silu_bf16(const float* __restrict__ xz,
                               const float* __restrict__ cw,
                               const float* __restrict__ cb,
                               __hip_bfloat16* __restrict__ xi, int rev)
{
    const int id = blockIdx.x * blockDim.x + threadIdx.x;
    if (id >= kBT * kDINNER) return;
    const int e = id % kDINNER;
    const int t = (id / kDINNER) % kT;
    const int b = id / (kDINNER * kT);
    float accv = cb[e];
#pragma unroll
    for (int k = 0; k < kDCONV; ++k) {
        const int j = rev ? (t + (kDCONV - 1) - k) : (t - (kDCONV - 1) + k);
        if (j >= 0 && j < kT)
            accv += cw[e * kDCONV + k] *
                    xz[(size_t)(b * kT + j) * (2 * kDINNER) + e];
    }
    const float s = accv / (1.0f + __expf(-accv));   // SiLU
    xi[(size_t)id] = __float2bfloat16(s);
}

// ---------------------------------------------------------------------------
// Slice dt = x_dbl[:, 0:64] and convert to bf16 (A input of delta GEMM)
// ---------------------------------------------------------------------------
__global__ void slice_dt_bf16(const float* __restrict__ xdbl,
                              __hip_bfloat16* __restrict__ dt)
{
    const int id = blockIdx.x * blockDim.x + threadIdx.x;
    if (id >= kBT * kDTRANK) return;
    const int r = id / kDTRANK, c = id % kDTRANK;
    dt[id] = __float2bfloat16(xdbl[(size_t)r * kXDBL + c]);
}

// ---------------------------------------------------------------------------
// Generic f32 -> bf16 (weight conversion)
// ---------------------------------------------------------------------------
__global__ void f32_to_bf16(const float* __restrict__ s,
                            __hip_bfloat16* __restrict__ d, int n)
{
    const int id = blockIdx.x * blockDim.x + threadIdx.x;
    if (id < n) d[id] = __float2bfloat16(s[id]);
}

// ---------------------------------------------------------------------------
// Selective scan, fused with D-skip and SiLU(z) gating; emits bf16 y
// (A input of the out-projection GEMM).  One lane per (b, d, state): 16 lanes
// cooperate per channel, reducing y with intra-16 shuffles.  rev=1 walks time
// backwards (equivalent to scanning the flipped sequence and flipping back).
// ---------------------------------------------------------------------------
__global__ __launch_bounds__(256)
void scan_fused(const float* __restrict__ delta,
                const __hip_bfloat16* __restrict__ xi,
                const float* __restrict__ xdbl,
                const float* __restrict__ xz,
                const float* __restrict__ A_log,
                const float* __restrict__ Dskip,
                __hip_bfloat16* __restrict__ ybf, int rev)
{
    const int n  = threadIdx.x & 15;                       // state index
    const int ch = blockIdx.x * (256 / 16) + (threadIdx.x >> 4);
    const int b  = ch / kDINNER;
    const int d  = ch % kDINNER;
    const float Av = -__expf(A_log[(size_t)d * kDSTATE + n]);
    const float Dv = Dskip[d];
    float h = 0.0f;
    for (int tt = 0; tt < kT; ++tt) {
        const int t = rev ? (kT - 1 - tt) : tt;
        const size_t row = (size_t)b * kT + t;
        const float dlt = delta[row * kDINNER + d];
        const float u   = __bfloat162float(xi[row * kDINNER + d]);
        const float Bv  = xdbl[row * kXDBL + kDTRANK + n];
        const float Cv  = xdbl[row * kXDBL + kDTRANK + kDSTATE + n];
        h = __expf(dlt * Av) * h + (dlt * u) * Bv;
        float y = h * Cv;
        y += __shfl_xor(y, 1, 16);
        y += __shfl_xor(y, 2, 16);
        y += __shfl_xor(y, 4, 16);
        y += __shfl_xor(y, 8, 16);
        if (n == 0) {
            const float z  = xz[row * (2 * kDINNER) + kDINNER + d];
            const float sz = z / (1.0f + __expf(-z));      // SiLU(z)
            ybf[row * kDINNER + d] = __float2bfloat16((y + u * Dv) * sz);
        }
    }
}

// ---------------------------------------------------------------------------
// Host orchestration
// ---------------------------------------------------------------------------
extern "C" void kernel_launch(void* const* d_in, const int* in_sizes, int n_in,
                              void* d_out, int out_size, void* d_ws, size_t ws_size,
                              hipStream_t stream)
{
    (void)in_sizes; (void)n_in; (void)out_size; (void)ws_size;

    const float* x       = (const float*)d_in[0];
    const float* ln_w    = (const float*)d_in[1];
    const float* ln_b    = (const float*)d_in[2];
    const float* comb_w  = (const float*)d_in[3];
    const float* comb_b  = (const float*)d_in[4];
    const unsigned char* mask = (const unsigned char*)d_in[5]; // jnp.bool_

    struct DirP { const float *in_w, *conv_w, *conv_b, *xproj_w, *dt_w,
                              *dt_b, *A_log, *D, *out_w; };
    DirP dir[2];
    for (int g = 0; g < 2; ++g) {
        const int o = 6 + g * 9;
        dir[g].in_w    = (const float*)d_in[o + 0];
        dir[g].conv_w  = (const float*)d_in[o + 1];
        dir[g].conv_b  = (const float*)d_in[o + 2];
        dir[g].xproj_w = (const float*)d_in[o + 3];
        dir[g].dt_w    = (const float*)d_in[o + 4];
        dir[g].dt_b    = (const float*)d_in[o + 5];
        dir[g].A_log   = (const float*)d_in[o + 6];
        dir[g].D       = (const float*)d_in[o + 7];
        dir[g].out_w   = (const float*)d_in[o + 8];
    }

    // ---- workspace layout (fixed offsets, 256B aligned) ----
    char* ws = (char*)d_ws;
    size_t off = 0;
    auto take = [&](size_t bytes) -> char* {
        char* p = ws + off;
        off += (bytes + 255) & ~size_t(255);
        return p;
    };
    typedef __hip_bfloat16 bf;
    bf* xn_bf      = (bf*)take((size_t)kBT * kDIM * 2);
    bf* inw_bf[2]  = { (bf*)take((size_t)2 * kDINNER * kDIM * 2),
                       (bf*)take((size_t)2 * kDINNER * kDIM * 2) };
    bf* xpw_bf[2]  = { (bf*)take((size_t)kXDBL * kDINNER * 2),
                       (bf*)take((size_t)kXDBL * kDINNER * 2) };
    bf* dtw_bf[2]  = { (bf*)take((size_t)kDINNER * kDTRANK * 2),
                       (bf*)take((size_t)kDINNER * kDTRANK * 2) };
    bf* outw_bf[2] = { (bf*)take((size_t)kDIM * kDINNER * 2),
                       (bf*)take((size_t)kDIM * kDINNER * 2) };
    bf* combw_bf   = (bf*)take((size_t)kDIM * (2 * kDIM) * 2);
    float* xz      = (float*)take((size_t)kBT * 2 * kDINNER * 4);   // reused per dir
    bf*    xi_bf   = (bf*)take((size_t)kBT * kDINNER * 2);          // reused
    float* xdbl    = (float*)take((size_t)kBT * kXDBL * 4);         // reused
    bf*    dt_bf   = (bf*)take((size_t)kBT * kDTRANK * 2);          // reused
    float* delta   = (float*)take((size_t)kBT * kDINNER * 4);       // reused
    bf*    y_bf    = (bf*)take((size_t)kBT * kDINNER * 2);          // reused
    bf*    cat_bf  = (bf*)take((size_t)kBT * 2 * kDIM * 2);         // fwd||bwd

    auto cvt = [&](const float* s, bf* d, int n) {
        f32_to_bf16<<<(n + 255) / 256, 256, 0, stream>>>(s, d, n);
    };
    auto gemm = [&](const bf* A, int lda, const bf* W, int ldw,
                    float* Cf, bf* Cbf, int ldc, int col_off,
                    int M, int N, int K, int mode,
                    const float* bias, const float* resid,
                    const unsigned char* msk) {
        dim3 g((N + 63) / 64, M / 128);
        gemm_bf16_wmma<<<g, 256, 0, stream>>>(A, lda, W, ldw, Cf, Cbf, ldc,
                                              col_off, M, N, K, mode, bias,
                                              resid, msk);
    };

    // ---- weight conversion (L2-resident thereafter) ----
    for (int g = 0; g < 2; ++g) {
        cvt(dir[g].in_w,    inw_bf[g],  2 * kDINNER * kDIM);
        cvt(dir[g].xproj_w, xpw_bf[g],  kXDBL * kDINNER);
        cvt(dir[g].dt_w,    dtw_bf[g],  kDINNER * kDTRANK);
        cvt(dir[g].out_w,   outw_bf[g], kDIM * kDINNER);
    }
    cvt(comb_w, combw_bf, kDIM * 2 * kDIM);

    // ---- LayerNorm -> bf16 tokens ----
    layernorm_bf16<<<kBT, 256, 0, stream>>>(x, ln_w, ln_b, xn_bf);

    // ---- two Mamba branches (bwd realized via reversed indexing) ----
    for (int g = 0; g < 2; ++g) {
        const int rev = g;  // g==1 is the time-flipped branch
        // in-projection: xz = xn @ in_w^T   [8192, 4096]
        gemm(xn_bf, kDIM, inw_bf[g], kDIM, xz, nullptr, 2 * kDINNER, 0,
             kBT, 2 * kDINNER, kDIM, 0, nullptr, nullptr, nullptr);
        // depthwise conv + SiLU -> xi (bf16)
        {
            const int n = kBT * kDINNER;
            conv_silu_bf16<<<(n + 255) / 256, 256, 0, stream>>>(
                xz, dir[g].conv_w, dir[g].conv_b, xi_bf, rev);
        }
        // x-projection: x_dbl = xi @ xproj_w^T   [8192, 96]
        gemm(xi_bf, kDINNER, xpw_bf[g], kDINNER, xdbl, nullptr, kXDBL, 0,
             kBT, kXDBL, kDINNER, 0, nullptr, nullptr, nullptr);
        // dt slice -> bf16
        {
            const int n = kBT * kDTRANK;
            slice_dt_bf16<<<(n + 255) / 256, 256, 0, stream>>>(xdbl, dt_bf);
        }
        // delta = softplus(dt @ dt_w^T + dt_b)   [8192, 2048]
        gemm(dt_bf, kDTRANK, dtw_bf[g], kDTRANK, delta, nullptr, kDINNER, 0,
             kBT, kDINNER, kDTRANK, 1, dir[g].dt_b, nullptr, nullptr);
        // selective scan fused with D-skip + SiLU(z) gating -> y_bf
        {
            const int channels = kBATCH * kDINNER;          // 8192
            scan_fused<<<channels / 16, 256, 0, stream>>>(
                delta, xi_bf, xdbl, xz, dir[g].A_log, dir[g].D, y_bf, rev);
        }
        // out-projection -> bf16 into its half of the concat buffer
        gemm(y_bf, kDINNER, outw_bf[g], kDINNER, nullptr, cat_bf,
             2 * kDIM, g * kDIM, kBT, kDIM, kDINNER, 2,
             nullptr, nullptr, nullptr);
    }

    // ---- combine + bias + residual + mask -> d_out (f32) ----
    gemm(cat_bf, 2 * kDIM, combw_bf, 2 * kDIM, (float*)d_out, nullptr,
         kDIM, 0, kBT, kDIM, 2 * kDIM, 3, comb_b, x, mask);
}
